// MultiomicTransformer_22084721836517
// MI455X (gfx1250) — matrix operand order repaired and checked by reading, not compiled, not run on hardware
//
#include <hip/hip_runtime.h>

// ---------------------------------------------------------------------------
// MultiomicTransformer for MI455X (gfx1250, wave32, WMMA).
// All matmuls -> v_wmma_f32_16x16x32_bf16 (bf16 in, f32 accum).
// LN / softmax / residual math in f32.
// GEMM: 256 thr = 8 wave32, block tile 128x64, wave tile 32x32 (2x2 WMMA),
// K-step 32, double-buffered LDS so next-tile global loads overlap WMMA.
// ---------------------------------------------------------------------------

typedef __bf16 bf16;
typedef __attribute__((ext_vector_type(8)))  bf16  bf16x8;
typedef __attribute__((ext_vector_type(16))) bf16  bf16x16;
typedef __attribute__((ext_vector_type(8)))  float f32x8;

#define B_    8
#define W_    2048
#define WIN_  1024
#define D_    512
#define NH_   8
#define HD_   64
#define DFF_  2048
#define L_    4
#define S_    512      // W_/KS
#define G_    2048
#define BS_   (B_*S_)  // 4096

#define MT 128
#define NT 64
#define KT 32
#define LDK (KT + 8)   // LDS row stride in bf16 (80B -> 16B aligned chunks)

// ------------------------------ GEMM (WMMA) --------------------------------
// C[z] = A[z](MxK) * B[z] + bias (+Res) ; B form: BL==0 -> B stored (N,K)
// row-major ("weight"/K-matrix form, computes A*B^T); BL==1 -> B stored (K,N).
// Requires: M%128==0, N%64==0, K%32==0, lda/ldb multiples of 8.
template<int BL, bool RELU, bool OUTBF, bool ADDRES>
__global__ __launch_bounds__(256)
void gemm_wmma(const bf16* __restrict__ A, const bf16* __restrict__ Bm,
               const float* __restrict__ bias, void* Cv, const float* Res,
               int M, int N, int K, int lda, int ldb, int ldc,
               long zsa, long zsb, long zsc, long zsr)
{
    __shared__ __align__(16) bf16 As[2][MT * LDK];
    __shared__ __align__(16) bf16 Bs[2][NT * LDK];
    const int tid = (int)threadIdx.x;
    const int z   = (int)blockIdx.z;
    const int m0  = (int)blockIdx.y * MT;
    const int n0  = (int)blockIdx.x * NT;
    const bf16* Ab = A  + (size_t)z * zsa;
    const bf16* Bb = Bm + (size_t)z * zsb;

    const int warp = tid >> 5, lane = tid & 31;
    const int wm = (warp >> 1) * 32, wn = (warp & 1) * 32;
    const int l0 = lane & 15, hi = lane >> 4;

    // tile-staging mappings (all 256 threads used)
    const int ar = tid >> 1, ac = (tid & 1) * 16;   // A: 128 rows x 32 cols
    const int br = tid >> 2, bc = (tid & 3) * 8;    // B (N,K): 64 rows x 32
    const int kr = tid >> 3, nc = (tid & 7) * 8;    // B (K,N): 32 rows x 64

    f32x8 acc[2][2] = {};
    bf16x8 ra0, ra1, rb0;

    // ---- prologue: stage K-tile 0 into LDS buffer 0 ----
    {
        const bf16* srcA = Ab + (size_t)(m0 + ar) * lda + ac;
        ra0 = *(const bf16x8*)srcA;
        ra1 = *(const bf16x8*)(srcA + 8);
        if (BL == 0) rb0 = *(const bf16x8*)(Bb + (size_t)(n0 + br) * ldb + bc);
        else         rb0 = *(const bf16x8*)(Bb + (size_t)kr * ldb + n0 + nc);
    }
    *(bf16x8*)&As[0][ar * LDK + ac]     = ra0;
    *(bf16x8*)&As[0][ar * LDK + ac + 8] = ra1;
    if (BL == 0) {
        *(bf16x8*)&Bs[0][br * LDK + bc] = rb0;
    } else {
        #pragma unroll
        for (int i = 0; i < 8; ++i) Bs[0][(nc + i) * LDK + kr] = rb0[i];
    }
    __syncthreads();

    int buf = 0;
    for (int k0 = 0; k0 < K; k0 += KT) {
        // next K-tile index (wraps to 0 on last step: branch-free dummy load)
        int kn = k0 + KT;
        if (kn >= K) kn = 0;

        // ---- issue next-tile global loads; they overlap the WMMAs below ----
        {
            const bf16* srcA = Ab + (size_t)(m0 + ar) * lda + kn + ac;
            ra0 = *(const bf16x8*)srcA;
            ra1 = *(const bf16x8*)(srcA + 8);
            __builtin_prefetch(srcA + KT, 0, 1);   // speculative, safe OOB
            if (BL == 0) {
                const bf16* srcB = Bb + (size_t)(n0 + br) * ldb + kn + bc;
                rb0 = *(const bf16x8*)srcB;
                __builtin_prefetch(srcB + KT, 0, 1);
            } else {
                const bf16* srcB = Bb + (size_t)(kn + kr) * ldb + n0 + nc;
                rb0 = *(const bf16x8*)srcB;
                __builtin_prefetch(srcB + (size_t)KT * ldb, 0, 1);
            }
        }

        // ---- fragments + WMMA from LDS[buf] ----
        // A (16x32): lane row = l0; VGPR0-3 hold K = hi*8..+8,
        //            VGPR4-7 hold K = 16+hi*8..+8 (two 16B chunks)
        // B (32x16): lane col = l0; same K split as A (two 16B chunks)
        bf16x16 af[2], bfm[2];
        #pragma unroll
        for (int i = 0; i < 2; ++i) {
            const bf16* pa = &As[buf][(wm + i*16 + l0) * LDK + hi * 8];
            bf16x8 alo = *(const bf16x8*)pa;
            bf16x8 ahi = *(const bf16x8*)(pa + 16);
            af[i] = __builtin_shufflevector(alo, ahi,
                     0,1,2,3,4,5,6,7,8,9,10,11,12,13,14,15);
            const bf16* pb = &Bs[buf][(wn + i*16 + l0) * LDK + hi * 8];
            bf16x8 blo = *(const bf16x8*)pb;
            bf16x8 bhi = *(const bf16x8*)(pb + 16);
            bfm[i] = __builtin_shufflevector(blo, bhi,
                     0,1,2,3,4,5,6,7,8,9,10,11,12,13,14,15);
        }
        #pragma unroll
        for (int i = 0; i < 2; ++i)
            #pragma unroll
            for (int j = 0; j < 2; ++j)
                acc[i][j] = __builtin_amdgcn_wmma_f32_16x16x32_bf16(
                    false, af[i], false, bfm[j], (short)0, acc[i][j],
                    false, false);

        // ---- stage next tile into the other LDS buffer ----
        const int nb = buf ^ 1;
        *(bf16x8*)&As[nb][ar * LDK + ac]     = ra0;
        *(bf16x8*)&As[nb][ar * LDK + ac + 8] = ra1;
        if (BL == 0) {
            *(bf16x8*)&Bs[nb][br * LDK + bc] = rb0;
        } else {
            #pragma unroll
            for (int i = 0; i < 8; ++i) Bs[nb][(nc + i) * LDK + kr] = rb0[i];
        }
        __syncthreads();
        buf = nb;
    }

    // epilogue: C/D layout -> VGPR r: lanes0-15 (M=r,N=l0), lanes16-31 (M=r+8)
    #pragma unroll
    for (int i = 0; i < 2; ++i) {
        #pragma unroll
        for (int j = 0; j < 2; ++j) {
            const int gmb = m0 + wm + i*16 + hi*8;
            const int gn  = n0 + wn + j*16 + l0;
            const float bv = bias ? bias[gn] : 0.f;
            #pragma unroll
            for (int r = 0; r < 8; ++r) {
                const int gm = gmb + r;
                float v = acc[i][j][r] + bv;
                if (ADDRES) v += Res[(size_t)z * zsr + (size_t)gm * ldc + gn];
                if (RELU)   v  = fmaxf(v, 0.f);
                const size_t idx = (size_t)z * zsc + (size_t)gm * ldc + gn;
                if (OUTBF) ((bf16*)Cv)[idx]  = (bf16)v;
                else       ((float*)Cv)[idx] = v;
            }
        }
    }
}

// ---------------------------- small kernels --------------------------------

__global__ void cast_bf16_kernel(const float* __restrict__ s,
                                 bf16* __restrict__ d, long n)
{
    long i = (long)blockIdx.x * 256 + threadIdx.x;
    if (i < n) d[i] = (bf16)s[i];
}

// mean-pool 4 window rows (commutes with linear projection) -> bf16
__global__ void pool_kernel(const float* __restrict__ win,
                            bf16* __restrict__ out)
{
    int i = blockIdx.x * 256 + threadIdx.x;   // over B_*S_*WIN_
    int c = i & (WIN_ - 1);
    int s = (i >> 10) & (S_ - 1);
    int b = i >> 19;
    const float* base = win + (((size_t)b * W_ + (size_t)s * 4) << 10) + c;
    float acc = base[0] + base[WIN_] + base[2*WIN_] + base[3*WIN_];
    out[i] = (bf16)(acc * 0.25f);
}

__global__ void posenc_kernel(float* __restrict__ x)
{
    int i = blockIdx.x * 256 + threadIdx.x;   // over BS_*D_
    int d = i & (D_ - 1);
    int s = (i >> 9) & (S_ - 1);
    int pi = d >> 1;
    float div = __expf((float)(2 * pi) * (-9.210340371976184f / (float)D_));
    float a = (float)s * div;
    x[i] += (d & 1) ? __cosf(a) : __sinf(a);
}

__global__ void gather_kernel(const float* __restrict__ ge,
                              const int* __restrict__ ids,
                              bf16* __restrict__ out)
{
    int i = blockIdx.x * 256 + threadIdx.x;   // over G_*D_
    int g = i >> 9, d = i & (D_ - 1);
    out[i] = (bf16)(ge[(size_t)ids[g] * D_ + d]);
}

__device__ __forceinline__ float wave_sum(float v)
{
    #pragma unroll
    for (int off = 16; off; off >>= 1) v += __shfl_xor(v, off, 32);
    return v;
}
__device__ __forceinline__ float wave_max(float v)
{
    #pragma unroll
    for (int off = 16; off; off >>= 1) v = fmaxf(v, __shfl_xor(v, off, 32));
    return v;
}

__global__ __launch_bounds__(256)
void ln_kernel(const float* __restrict__ X, const float* __restrict__ g,
               const float* __restrict__ b, bf16* __restrict__ Y)
{
    const int row = blockIdx.x, tid = threadIdx.x;
    const float* x = X + (size_t)row * D_;
    float v0 = x[tid], v1 = x[tid + 256];
    float s  = wave_sum(v0 + v1);
    float ss = wave_sum(v0*v0 + v1*v1);
    __shared__ float sm[16];
    if ((tid & 31) == 0) { sm[tid >> 5] = s; sm[8 + (tid >> 5)] = ss; }
    __syncthreads();
    float ts = 0.f, tss = 0.f;
    #pragma unroll
    for (int i = 0; i < 8; ++i) { ts += sm[i]; tss += sm[8 + i]; }
    float m  = ts * (1.f / D_);
    float rs = rsqrtf(tss * (1.f / D_) - m * m + 1e-5f);
    bf16* y = Y + (size_t)row * D_;
    y[tid]       = (bf16)((v0 - m) * rs * g[tid]       + b[tid]);
    y[tid + 256] = (bf16)((v1 - m) * rs * g[tid + 256] + b[tid + 256]);
}

__global__ __launch_bounds__(256)
void softmax_kernel(const float* __restrict__ S, bf16* __restrict__ P,
                    float scale)
{
    const int row = blockIdx.x, tid = threadIdx.x;  // 512 cols
    const float* s = S + (size_t)row * 512;
    float v0 = s[tid] * scale, v1 = s[tid + 256] * scale;
    float mx = wave_max(fmaxf(v0, v1));
    __shared__ float sm[16];
    if ((tid & 31) == 0) sm[tid >> 5] = mx;
    __syncthreads();
    float m = sm[0];
    #pragma unroll
    for (int i = 1; i < 8; ++i) m = fmaxf(m, sm[i]);
    float e0 = __expf(v0 - m), e1 = __expf(v1 - m);
    float su = wave_sum(e0 + e1);
    if ((tid & 31) == 0) sm[8 + (tid >> 5)] = su;
    __syncthreads();
    float tot = 0.f;
    #pragma unroll
    for (int i = 0; i < 8; ++i) tot += sm[8 + i];
    float inv = 1.f / tot;
    bf16* p = P + (size_t)row * 512;
    p[tid]       = (bf16)(e0 * inv);
    p[tid + 256] = (bf16)(e1 * inv);
}

__global__ __launch_bounds__(256)
void readout_kernel(const float* __restrict__ Z, const float* __restrict__ g,
                    const float* __restrict__ b, const float* __restrict__ w,
                    const float* __restrict__ b0, float* __restrict__ pred)
{
    const int row = blockIdx.x, tid = threadIdx.x;
    const float* z = Z + (size_t)row * D_;
    float v0 = z[tid], v1 = z[tid + 256];
    float s  = wave_sum(v0 + v1);
    float ss = wave_sum(v0*v0 + v1*v1);
    __shared__ float sm[24];
    if ((tid & 31) == 0) { sm[tid >> 5] = s; sm[8 + (tid >> 5)] = ss; }
    __syncthreads();
    float ts = 0.f, tss = 0.f;
    #pragma unroll
    for (int i = 0; i < 8; ++i) { ts += sm[i]; tss += sm[8 + i]; }
    float m  = ts * (1.f / D_);
    float rs = rsqrtf(tss * (1.f / D_) - m * m + 1e-5f);
    float d = ((v0 - m) * rs * g[tid]       + b[tid])       * w[tid]
            + ((v1 - m) * rs * g[tid + 256] + b[tid + 256]) * w[tid + 256];
    d = wave_sum(d);
    if ((tid & 31) == 0) sm[16 + (tid >> 5)] = d;
    __syncthreads();
    if (tid == 0) {
        float tot = 0.f;
        #pragma unroll
        for (int i = 0; i < 8; ++i) tot += sm[16 + i];
        pred[row] = tot + b0[0];
    }
}

// ------------------------------ host side ----------------------------------

static inline char* wsalloc(char*& p, size_t bytes)
{
    char* r = p;
    p += (bytes + 255) & ~(size_t)255;
    return r;
}

extern "C" void kernel_launch(void* const* d_in, const int* in_sizes, int n_in,
                              void* d_out, int out_size, void* d_ws,
                              size_t ws_size, hipStream_t stream)
{
    const float* windows   = (const float*)d_in[0];
    const int*   gene_ids  = (const int*)  d_in[1];
    const float* W_window  = (const float*)d_in[2];
    const float* enc_qkv_w = (const float*)d_in[3];
    const float* enc_qkv_b = (const float*)d_in[4];
    const float* enc_out_w = (const float*)d_in[5];
    const float* enc_out_b = (const float*)d_in[6];
    const float* ln1g = (const float*)d_in[7];
    const float* ln1b = (const float*)d_in[8];
    const float* ln2g = (const float*)d_in[9];
    const float* ln2b = (const float*)d_in[10];
    const float* ff1w = (const float*)d_in[11];
    const float* ff1b = (const float*)d_in[12];
    const float* ff2w = (const float*)d_in[13];
    const float* ff2b = (const float*)d_in[14];
    const float* gene_embed = (const float*)d_in[15];
    const float* ca_qkv_w   = (const float*)d_in[16];
    const float* ca_qkv_b   = (const float*)d_in[17];
    const float* ca_out_w   = (const float*)d_in[18];
    const float* ca_out_b   = (const float*)d_in[19];
    const float* ro_ln_g = (const float*)d_in[20];
    const float* ro_ln_b = (const float*)d_in[21];
    const float* ro_w    = (const float*)d_in[22];
    const float* ro_b    = (const float*)d_in[23];
    float* out = (float*)d_out;

    // ---- workspace carve ----
    char* p = (char*)d_ws;
    bf16* wwin_bf  = (bf16*)wsalloc(p, (size_t)D_*WIN_       * 2);
    bf16* qkvw_bf  = (bf16*)wsalloc(p, (size_t)L_*3*D_*D_    * 2);
    bf16* outw_bf  = (bf16*)wsalloc(p, (size_t)L_*D_*D_      * 2);
    bf16* ff1w_bf  = (bf16*)wsalloc(p, (size_t)L_*DFF_*D_    * 2);
    bf16* ff2w_bf  = (bf16*)wsalloc(p, (size_t)L_*D_*DFF_    * 2);
    bf16* caqkv_bf = (bf16*)wsalloc(p, (size_t)3*D_*D_       * 2);
    bf16* caout_bf = (bf16*)wsalloc(p, (size_t)D_*D_         * 2);
    bf16* pooled_bf= (bf16*)wsalloc(p, (size_t)BS_*WIN_      * 2);
    bf16* h_bf     = (bf16*)wsalloc(p, (size_t)BS_*D_        * 2);
    bf16* qkv_bf   = (bf16*)wsalloc(p, (size_t)BS_*3*D_      * 2);
    bf16* attn_bf  = (bf16*)wsalloc(p, (size_t)BS_*D_        * 2);
    bf16* ff_bf    = (bf16*)wsalloc(p, (size_t)BS_*DFF_      * 2);
    bf16* xbf      = (bf16*)wsalloc(p, (size_t)BS_*D_        * 2);
    bf16* qin_bf   = (bf16*)wsalloc(p, (size_t)G_*D_         * 2);
    bf16* qg_bf    = (bf16*)wsalloc(p, (size_t)G_*D_         * 2);
    bf16* kv_bf    = (bf16*)wsalloc(p, (size_t)BS_*2*D_      * 2);
    bf16* probs_bf = (bf16*)wsalloc(p, (size_t)NH_*G_*S_     * 2);
    bf16* zattn_bf = (bf16*)wsalloc(p, (size_t)G_*D_         * 2);
    float* x       = (float*)wsalloc(p, (size_t)BS_*D_       * 4);
    float* scoresf = (float*)wsalloc(p, (size_t)NH_*G_*S_    * 4);
    (void)ws_size; (void)n_in; (void)in_sizes; (void)out_size;

    auto cast = [&](const float* s, bf16* d, long n) {
        cast_bf16_kernel<<<(n + 255) / 256, 256, 0, stream>>>(s, d, n);
    };

    // ---- weights -> bf16 ----
    cast(W_window,  wwin_bf,  (long)D_ * WIN_);
    cast(enc_qkv_w, qkvw_bf,  (long)L_ * 3 * D_ * D_);
    cast(enc_out_w, outw_bf,  (long)L_ * D_ * D_);
    cast(ff1w,      ff1w_bf,  (long)L_ * DFF_ * D_);
    cast(ff2w,      ff2w_bf,  (long)L_ * D_ * DFF_);
    cast(ca_qkv_w,  caqkv_bf, (long)3 * D_ * D_);
    cast(ca_out_w,  caout_bf, (long)D_ * D_);

    // ---- stem: pool -> project -> +posenc ----
    pool_kernel<<<(BS_ * WIN_) / 256, 256, 0, stream>>>(windows, pooled_bf);
    gemm_wmma<0,false,false,false><<<dim3(D_/NT, BS_/MT, 1), 256, 0, stream>>>(
        pooled_bf, wwin_bf, nullptr, x, nullptr,
        BS_, D_, WIN_, WIN_, WIN_, D_, 0, 0, 0, 0);
    posenc_kernel<<<(BS_ * D_) / 256, 256, 0, stream>>>(x);

    // ---- encoder layers ----
    for (int i = 0; i < L_; ++i) {
        ln_kernel<<<BS_, 256, 0, stream>>>(x, ln1g + i*D_, ln1b + i*D_, h_bf);
        // qkv = h @ Wqkv^T + b : (4096 x 1536)
        gemm_wmma<0,false,true,false><<<dim3(3*D_/NT, BS_/MT, 1), 256, 0, stream>>>(
            h_bf, qkvw_bf + (size_t)i*3*D_*D_, enc_qkv_b + i*3*D_, qkv_bf, nullptr,
            BS_, 3*D_, D_, D_, D_, 3*D_, 0, 0, 0, 0);
        for (int b = 0; b < B_; ++b) {
            const bf16* qkvB = qkv_bf + (size_t)b * S_ * 3 * D_;
            // scores[h] = Q_h (512x64) @ K_h^T : per-head via grid.z
            gemm_wmma<0,false,false,false><<<dim3(S_/NT, S_/MT, NH_), 256, 0, stream>>>(
                qkvB, qkvB + D_, nullptr, scoresf, nullptr,
                S_, S_, HD_, 3*D_, 3*D_, S_, HD_, HD_, (long)S_*S_, 0);
            softmax_kernel<<<NH_ * S_, 256, 0, stream>>>(scoresf, probs_bf, 0.125f);
            // attn = P (512x512) @ V_h (512x64), bf16 out, head-interleaved
            gemm_wmma<1,false,true,false><<<dim3(HD_/NT, S_/MT, NH_), 256, 0, stream>>>(
                probs_bf, qkvB + 2*D_, nullptr,
                attn_bf + (size_t)b * S_ * D_, nullptr,
                S_, HD_, S_, S_, 3*D_, D_, (long)S_*S_, HD_, HD_, 0);
        }
        // x += attn @ Wout^T + b
        gemm_wmma<0,false,false,true><<<dim3(D_/NT, BS_/MT, 1), 256, 0, stream>>>(
            attn_bf, outw_bf + (size_t)i*D_*D_, enc_out_b + i*D_, x, x,
            BS_, D_, D_, D_, D_, D_, 0, 0, 0, 0);
        // FFN
        ln_kernel<<<BS_, 256, 0, stream>>>(x, ln2g + i*D_, ln2b + i*D_, h_bf);
        gemm_wmma<0,true,true,false><<<dim3(DFF_/NT, BS_/MT, 1), 256, 0, stream>>>(
            h_bf, ff1w_bf + (size_t)i*DFF_*D_, ff1b + i*DFF_, ff_bf, nullptr,
            BS_, DFF_, D_, D_, D_, DFF_, 0, 0, 0, 0);
        gemm_wmma<0,false,false,true><<<dim3(D_/NT, BS_/MT, 1), 256, 0, stream>>>(
            ff_bf, ff2w_bf + (size_t)i*D_*DFF_, ff2b + i*D_, x, x,
            BS_, D_, DFF_, DFF_, DFF_, D_, 0, 0, 0, 0);
    }

    // ---- cross attention ----
    cast(x, xbf, (long)BS_ * D_);
    // K,V = x @ Wkv^T + b  (4096 x 1024)
    gemm_wmma<0,false,true,false><<<dim3(2*D_/NT, BS_/MT, 1), 256, 0, stream>>>(
        xbf, caqkv_bf + (size_t)D_*D_, ca_qkv_b + D_, kv_bf, nullptr,
        BS_, 2*D_, D_, D_, D_, 2*D_, 0, 0, 0, 0);
    // Q = gene_embed[ids] @ Wq^T + b  (shared across batches)
    gather_kernel<<<(G_ * D_) / 256, 256, 0, stream>>>(gene_embed, gene_ids, qin_bf);
    gemm_wmma<0,false,true,false><<<dim3(D_/NT, G_/MT, 1), 256, 0, stream>>>(
        qin_bf, caqkv_bf, ca_qkv_b, qg_bf, nullptr,
        G_, D_, D_, D_, D_, D_, 0, 0, 0, 0);

    float* zout = out + (size_t)B_ * G_;   // z region of d_out
    for (int b = 0; b < B_; ++b) {
        const bf16* kvB = kv_bf + (size_t)b * S_ * 2 * D_;
        gemm_wmma<0,false,false,false><<<dim3(S_/NT, G_/MT, NH_), 256, 0, stream>>>(
            qg_bf, kvB, nullptr, scoresf, nullptr,
            G_, S_, HD_, D_, 2*D_, S_, HD_, HD_, (long)G_*S_, 0);
        softmax_kernel<<<NH_ * G_, 256, 0, stream>>>(scoresf, probs_bf, 0.125f);
        gemm_wmma<1,false,true,false><<<dim3(HD_/NT, G_/MT, NH_), 256, 0, stream>>>(
            probs_bf, kvB + D_, nullptr, zattn_bf, nullptr,
            G_, HD_, S_, S_, 2*D_, D_, (long)G_*S_, HD_, HD_, 0);
        // z_b = attn @ Wout^T + b  -> straight into d_out (f32)
        gemm_wmma<0,false,false,false><<<dim3(D_/NT, G_/MT, 1), 256, 0, stream>>>(
            zattn_bf, caout_bf, ca_out_b, zout + (size_t)b * G_ * D_, nullptr,
            G_, D_, D_, D_, D_, D_, 0, 0, 0, 0);
    }

    // ---- readout: pred = LN(z) @ ro_w^T + ro_b ----
    readout_kernel<<<B_ * G_, 256, 0, stream>>>(zout, ro_ln_g, ro_ln_b,
                                                ro_w, ro_b, out);
}
